// PointnetMSG_17961553232576
// MI455X (gfx1250) — matrix operand order repaired and checked by reading, not compiled, not run on hardware
//
#include <hip/hip_runtime.h>
#include <hip/hip_bf16.h>
#include <math.h>

// ---------------- problem constants (from reference) ----------------
#define BATCH 8
#define NPTS  4096
#define NCLS  13
#define EPS_BN 1e-5f

typedef __attribute__((ext_vector_type(16))) _Float16 v16h;
typedef __attribute__((ext_vector_type(8)))  _Float16 v8h;
typedef __attribute__((ext_vector_type(8)))  float    v8f;

static __host__ __device__ inline int pad32(int x) { return (x + 31) & ~31; }

// =====================================================================
// transpose xyz (B,3,N) -> xyz_t (B,N,3)
// =====================================================================
__global__ void transpose_xyz_kernel(const float* __restrict__ src,
                                     float* __restrict__ dst) {
  int gid = blockIdx.x * blockDim.x + threadIdx.x;
  if (gid >= BATCH * NPTS) return;
  int b = gid / NPTS, p = gid % NPTS;
  const float* s = src + (size_t)b * 3 * NPTS;
  float* d = dst + ((size_t)b * NPTS + p) * 3;
  d[0] = s[0 * NPTS + p];
  d[1] = s[1 * NPTS + p];
  d[2] = s[2 * NPTS + p];
}

// =====================================================================
// Farthest point sampling: one block per batch, LDS-resident distances
// =====================================================================
#define FPS_BLOCK 256
__global__ void fps_kernel(const float* __restrict__ xyz, // (B,N,3)
                           int* __restrict__ idx_out,     // (B,npoint)
                           int N, int npoint) {
  __shared__ float dist[4096];
  __shared__ float rv[FPS_BLOCK];
  __shared__ int   ri[FPS_BLOCK];
  int b = blockIdx.x;
  int t = threadIdx.x;
  const float* px = xyz + (size_t)b * N * 3;
  for (int i = t; i < N; i += FPS_BLOCK) dist[i] = 1e10f;
  __syncthreads();
  int far = 0;
  for (int it = 0; it < npoint; ++it) {
    if (t == 0) idx_out[(size_t)b * npoint + it] = far;
    float cx = px[far * 3 + 0], cy = px[far * 3 + 1], cz = px[far * 3 + 2];
    float best = -1.0f; int bi = 0;
    for (int i = t; i < N; i += FPS_BLOCK) {
      float dx = px[i * 3 + 0] - cx;
      float dy = px[i * 3 + 1] - cy;
      float dz = px[i * 3 + 2] - cz;
      float d = dx * dx + dy * dy + dz * dz;
      float nd = fminf(dist[i], d);
      dist[i] = nd;
      if (nd > best) { best = nd; bi = i; }
    }
    rv[t] = best; ri[t] = bi;
    __syncthreads();
    for (int s = FPS_BLOCK / 2; s > 0; s >>= 1) {
      if (t < s) {
        if (rv[t + s] > rv[t] || (rv[t + s] == rv[t] && ri[t + s] < ri[t])) {
          rv[t] = rv[t + s]; ri[t] = ri[t + s];
        }
      }
      __syncthreads();
    }
    far = ri[0];
    __syncthreads();
  }
}

// =====================================================================
// gather sampled centers: new_xyz[b,p,:] = xyz[b, idx[b,p], :]
// =====================================================================
__global__ void gather_xyz_kernel(const float* __restrict__ xyz,
                                  const int* __restrict__ idx,
                                  float* __restrict__ out, int N, int Np) {
  int gid = blockIdx.x * blockDim.x + threadIdx.x;
  if (gid >= BATCH * Np) return;
  int b = gid / Np;
  int j = idx[gid];
  const float* s = xyz + ((size_t)b * N + j) * 3;
  float* d = out + (size_t)gid * 3;
  d[0] = s[0]; d[1] = s[1]; d[2] = s[2];
}

// =====================================================================
// ball query: first K indices (ascending) within radius, pad with first
// =====================================================================
__global__ void ball_query_kernel(const float* __restrict__ xyz,
                                  const float* __restrict__ nxyz,
                                  int* __restrict__ idx,
                                  int N, int Np, int K, float r2) {
  int gid = blockIdx.x * blockDim.x + threadIdx.x;
  if (gid >= BATCH * Np) return;
  int b = gid / Np;
  const float* px = xyz + (size_t)b * N * 3;
  float cx = nxyz[(size_t)gid * 3 + 0];
  float cy = nxyz[(size_t)gid * 3 + 1];
  float cz = nxyz[(size_t)gid * 3 + 2];
  int* out = idx + (size_t)gid * K;
  int cnt = 0, first = 0;
  for (int j = 0; j < N && cnt < K; ++j) {
    float dx = px[j * 3 + 0] - cx;
    float dy = px[j * 3 + 1] - cy;
    float dz = px[j * 3 + 2] - cz;
    float d = dx * dx + dy * dy + dz * dz;
    if (d <= r2) {
      if (cnt == 0) first = j;
      out[cnt++] = j;
    }
  }
  for (; cnt < K; ++cnt) out[cnt] = first;
}

// =====================================================================
// grouping into fp16 with zero-padded channel stride Cgp:
//   g[b,p,k,:] = concat(points[b,idx], xyz[b,idx]-new_xyz[b,p], zeros)
// =====================================================================
__global__ void group_kernel(const float* __restrict__ xyz,
                             const float* __restrict__ nxyz,
                             const _Float16* __restrict__ pts, int Cp,
                             const int* __restrict__ idx,
                             _Float16* __restrict__ g,
                             int N, int Np, int K, int Cgp) {
  int gid = blockIdx.x * blockDim.x + threadIdx.x;
  int M = BATCH * Np * K;
  if (gid >= M) return;
  int bp = gid / K;
  int b = bp / Np;
  int j = idx[gid];
  _Float16* out = g + (size_t)gid * Cgp;
  if (pts) {
    const _Float16* s = pts + ((size_t)b * N + j) * Cp;
    for (int c = 0; c < Cp; ++c) out[c] = s[c];
  }
  const float* pxj = xyz + ((size_t)b * N + j) * 3;
  const float* cen = nxyz + (size_t)bp * 3;
  out[Cp + 0] = (_Float16)(pxj[0] - cen[0]);
  out[Cp + 1] = (_Float16)(pxj[1] - cen[1]);
  out[Cp + 2] = (_Float16)(pxj[2] - cen[2]);
  for (int c = Cp + 3; c < Cgp; ++c) out[c] = (_Float16)0.0f;
}

// =====================================================================
// stage weights: fp32 (Nlog,Klog) -> fp16 zero-padded (Npad,Kpad)
// =====================================================================
__global__ void pad_w_kernel(const float* __restrict__ W,
                             _Float16* __restrict__ Wp,
                             int Nlog, int Npad, int Klog, int Kpad) {
  int gid = blockIdx.x * blockDim.x + threadIdx.x;
  int total = Npad * Kpad;
  if (gid >= total) return;
  int n = gid / Kpad, k = gid % Kpad;
  Wp[gid] = (n < Nlog && k < Klog) ? (_Float16)W[(size_t)n * Klog + k]
                                   : (_Float16)0.0f;
}

// =====================================================================
// WMMA linear: Y[m,n] = sum_k X[m,k]*W[n,k]  (+bias), fp16 in, fp32 out
// X: (M,Ks) fp16, Ks%32==0, rows 16B-aligned. Wp: (Npad,Ks) fp16 staged.
// One wave per 16x16 tile. No guards in the K loop: 4x b128 load + wmma.
// Fragment layouts per CDNA5 ISA 7.12.2 (wave32).
// =====================================================================
__global__ void linear_wmma_kernel(const _Float16* __restrict__ X,
                                   const _Float16* __restrict__ Wp,
                                   const float* __restrict__ bias,
                                   float* __restrict__ Y,
                                   int M, int Nlog, int Npad, int Ks) {
  int wave = (blockIdx.x * blockDim.x + threadIdx.x) >> 5;
  int lane = threadIdx.x & 31;
  int tiles_n = Npad >> 4;
  int tm = wave / tiles_n;
  int tn = wave - tm * tiles_n;
  if (tm * 16 >= M) return;
  int half_id = lane >> 4;   // 0: lanes 0-15, 1: lanes 16-31
  int l16 = lane & 15;
  int arow = tm * 16 + l16;  // A: M = lane%16; K sub-base = half_id*8
  int bcol = tn * 16 + l16;  // B/C/D: N = lane%16; B K sub-base = half_id*8
  const v8h* pa = (const v8h*)(X + (size_t)arow * Ks + half_id * 8);
  const v8h* pb = (const v8h*)(Wp + (size_t)bcol * Ks + half_id * 8);
  v8f c = {};
  int chunks = Ks >> 5;
  for (int i = 0; i < chunks; ++i) {
    v8h a0 = pa[0];            // K offs  kb+0..7
    v8h a1 = pa[2];            // K offs  kb+16..23
    v8h b0 = pb[0];            // K offs  koff+0..7
    v8h b1 = pb[2];            // K offs  koff+16..23
    v16h a = __builtin_shufflevector(a0, a1, 0, 1, 2, 3, 4, 5, 6, 7,
                                     8, 9, 10, 11, 12, 13, 14, 15);
    v16h bf = __builtin_shufflevector(b0, b1, 0, 1, 2, 3, 4, 5, 6, 7,
                                      8, 9, 10, 11, 12, 13, 14, 15);
    c = __builtin_amdgcn_wmma_f32_16x16x32_f16(
        /*neg_a=*/false, a, /*neg_b=*/false, bf,
        /*c_mod=*/(short)0, c, /*reuse_a=*/false, /*reuse_b=*/false);
    pa += 4;  // advance 32 halfs
    pb += 4;
  }
  float bv = (bcol < Nlog) ? bias[bcol] : 0.0f;
  int mbase = tm * 16 + half_id * 8;  // C/D: lanes 16-31 hold M=8..15
#pragma unroll
  for (int r = 0; r < 8; ++r) {
    Y[(size_t)(mbase + r) * Npad + bcol] = c[r] + bv;
  }
}

// =====================================================================
// BatchNorm stats: one block per channel -> stats[2c]=mean, [2c+1]=var
// =====================================================================
__global__ void bn_stats_kernel(const float* __restrict__ Y,
                                float* __restrict__ stats, int M, int N) {
  __shared__ float s1[256], s2[256];
  int ch = blockIdx.x, t = threadIdx.x;
  float a = 0.0f, b = 0.0f;
  for (int m = t; m < M; m += 256) {
    float v = Y[(size_t)m * N + ch];
    a += v; b += v * v;
  }
  s1[t] = a; s2[t] = b;
  __syncthreads();
  for (int s = 128; s > 0; s >>= 1) {
    if (t < s) { s1[t] += s1[t + s]; s2[t] += s2[t + s]; }
    __syncthreads();
  }
  if (t == 0) {
    float mean = s1[0] / (float)M;
    float var  = s2[0] / (float)M - mean * mean;
    stats[2 * ch]     = mean;
    stats[2 * ch + 1] = var;
  }
}

// =====================================================================
// BN normalize + ReLU: f32 Y -> fp16 activations (pad channels -> 0)
// =====================================================================
__global__ void bn_relu_kernel(const float* __restrict__ Y,
                               const float* __restrict__ stats,
                               const float* __restrict__ gamma,
                               const float* __restrict__ beta,
                               _Float16* __restrict__ dst,
                               int total, int Npad, int Nlog) {
  int gid = blockIdx.x * blockDim.x + threadIdx.x;
  if (gid >= total) return;
  int ch = gid % Npad;
  float r = 0.0f;
  if (ch < Nlog) {
    float m = stats[2 * ch], v = stats[2 * ch + 1];
    float y = gamma[ch] * (Y[gid] - m) * rsqrtf(v + EPS_BN) + beta[ch];
    r = fmaxf(y, 0.0f);
  }
  dst[gid] = (_Float16)r;
}

// =====================================================================
// Max-pool over K samples (fp16 in, stride Ns) into fp16 concat buffer
// =====================================================================
__global__ void maxpool_kernel(const _Float16* __restrict__ Y,
                               _Float16* __restrict__ out,
                               int Np, int K, int C, int Ns, int Cout, int coff) {
  int gid = blockIdx.x * blockDim.x + threadIdx.x;
  int total = BATCH * Np * C;
  if (gid >= total) return;
  int bp = gid / C, d = gid % C;
  float m = -1e30f;
  const _Float16* src = Y + ((size_t)bp * K) * Ns + d;
  for (int k = 0; k < K; ++k) m = fmaxf(m, (float)src[(size_t)k * Ns]);
  out[(size_t)bp * Cout + coff + d] = (_Float16)m;
}

// =====================================================================
// channel-sliced fp16 copy: dst[r, coff+c] = src[r, c]
// =====================================================================
__global__ void copy_ch_kernel(const _Float16* __restrict__ src, int Cin,
                               _Float16* __restrict__ dst, int Cout, int coff,
                               int rows) {
  int gid = blockIdx.x * blockDim.x + threadIdx.x;
  int total = rows * Cin;
  if (gid >= total) return;
  int r = gid / Cin, c = gid % Cin;
  dst[(size_t)r * Cout + coff + c] = src[(size_t)r * Cin + c];
}

// =====================================================================
// 3-NN inverse-distance interpolation into fp16 concat slot
// =====================================================================
__global__ void three_interp_kernel(const float* __restrict__ x1,
                                    const float* __restrict__ x2,
                                    const _Float16* __restrict__ p2,
                                    _Float16* __restrict__ dst,
                                    int N1, int N2, int C2, int Cout, int coff) {
  int gid = blockIdx.x * blockDim.x + threadIdx.x;
  if (gid >= BATCH * N1) return;
  int b = gid / N1;
  float cx = x1[(size_t)gid * 3 + 0];
  float cy = x1[(size_t)gid * 3 + 1];
  float cz = x1[(size_t)gid * 3 + 2];
  const float* px = x2 + (size_t)b * N2 * 3;
  float d0 = 1e30f, d1 = 1e30f, d2 = 1e30f;
  int i0 = 0, i1 = 0, i2 = 0;
  for (int j = 0; j < N2; ++j) {
    float dx = px[j * 3 + 0] - cx;
    float dy = px[j * 3 + 1] - cy;
    float dz = px[j * 3 + 2] - cz;
    float d = dx * dx + dy * dy + dz * dz;
    if (d < d0)      { d2 = d1; i2 = i1; d1 = d0; i1 = i0; d0 = d; i0 = j; }
    else if (d < d1) { d2 = d1; i2 = i1; d1 = d; i1 = j; }
    else if (d < d2) { d2 = d; i2 = j; }
  }
  float w0 = 1.0f / (d0 + 1e-8f);
  float w1 = 1.0f / (d1 + 1e-8f);
  float w2 = 1.0f / (d2 + 1e-8f);
  float ws = w0 + w1 + w2;
  w0 /= ws; w1 /= ws; w2 /= ws;
  const _Float16* f0 = p2 + ((size_t)b * N2 + i0) * C2;
  const _Float16* f1 = p2 + ((size_t)b * N2 + i1) * C2;
  const _Float16* f2 = p2 + ((size_t)b * N2 + i2) * C2;
  _Float16* out = dst + (size_t)gid * Cout + coff;
  for (int c = 0; c < C2; ++c)
    out[c] = (_Float16)(w0 * (float)f0[c] + w1 * (float)f1[c] + w2 * (float)f2[c]);
}

// =====================================================================
// final classifier: out[row, cls] = sum_c h[row,c]*W[cls,c] + b[cls]
// =====================================================================
__global__ void final_conv_kernel(const _Float16* __restrict__ h,
                                  const float* __restrict__ W,
                                  const float* __restrict__ bias,
                                  float* __restrict__ out) {
  int gid = blockIdx.x * blockDim.x + threadIdx.x;
  int total = BATCH * NPTS * NCLS;
  if (gid >= total) return;
  int row = gid / NCLS, cls = gid % NCLS;
  const _Float16* hr = h + (size_t)row * 128;
  const float* wr = W + (size_t)cls * 128;
  float acc = bias[cls];
  for (int c = 0; c < 128; ++c) acc += (float)hr[c] * wr[c];
  out[gid] = acc;
}

// =====================================================================
// Host-side orchestration
// =====================================================================
struct LayerEx {
  const float *W, *b, *gamma, *beta;
  int Nlog, Npad, Klog, Kpad;
  _Float16* Wp;
};

static inline unsigned cdivu(long long a, int b) { return (unsigned)((a + b - 1) / b); }

// Linear+BN+ReLU chain over fp16 ping-pong buffers; returns final fp16 ptr
static const _Float16* mlp_chain(const _Float16* X, int M, int Ks0,
                                 const LayerEx* Ls, int nL,
                                 _Float16* hA, _Float16* hB, float* yscr,
                                 float* stats, hipStream_t st) {
  const _Float16* cur = X;
  int Ks = Ks0;
  for (int i = 0; i < nL; ++i) {
    const LayerEx& L = Ls[i];
    _Float16* dst = (cur == hA) ? hB : hA;
    unsigned waves = (unsigned)((M / 16) * (L.Npad / 16));
    linear_wmma_kernel<<<cdivu((long long)waves * 32, 256), 256, 0, st>>>(
        cur, L.Wp, L.b, yscr, M, L.Nlog, L.Npad, Ks);
    bn_stats_kernel<<<L.Npad, 256, 0, st>>>(yscr, stats, M, L.Npad);
    int total = M * L.Npad;
    bn_relu_kernel<<<cdivu(total, 256), 256, 0, st>>>(
        yscr, stats, L.gamma, L.beta, dst, total, L.Npad, L.Nlog);
    cur = dst; Ks = L.Npad;
  }
  return cur;
}

static void sa_branch(const float* xyz, int N, const float* nxyz, int Np,
                      const _Float16* pts, int Cp, float radius, int Ks,
                      const LayerEx* Ls, int nL,
                      int* idxbuf, _Float16* hA, _Float16* hB, float* yscr,
                      float* stats, _Float16* cat, int Ctot, int coff,
                      hipStream_t st) {
  int nctr = BATCH * Np;
  ball_query_kernel<<<cdivu(nctr, 256), 256, 0, st>>>(
      xyz, nxyz, idxbuf, N, Np, Ks, radius * radius);
  int M = BATCH * Np * Ks;
  int Cgp = pad32(Cp + 3);
  group_kernel<<<cdivu(M, 256), 256, 0, st>>>(
      xyz, nxyz, pts, Cp, idxbuf, hA, N, Np, Ks, Cgp);
  const _Float16* y = mlp_chain(hA, M, Cgp, Ls, nL, hA, hB, yscr, stats, st);
  int outC = Ls[nL - 1].Nlog, outS = Ls[nL - 1].Npad;
  maxpool_kernel<<<cdivu((long long)BATCH * Np * outC, 256), 256, 0, st>>>(
      y, cat, Np, Ks, outC, outS, Ctot, coff);
}

static const _Float16* fp_stage(const float* x1, int N1, const float* x2, int N2,
                                const _Float16* p1, int C1,
                                const _Float16* p2, int C2,
                                const LayerEx* Ls, int nL,
                                _Float16* hA, _Float16* hB, float* yscr,
                                float* stats, hipStream_t st) {
  int Ctot = C1 + C2;  // always a multiple of 32 here
  int rows = BATCH * N1;
  if (p1)
    copy_ch_kernel<<<cdivu((long long)rows * C1, 256), 256, 0, st>>>(
        p1, C1, hA, Ctot, 0, rows);
  three_interp_kernel<<<cdivu(rows, 256), 256, 0, st>>>(
      x1, x2, p2, hA, N1, N2, C2, Ctot, C1);
  return mlp_chain(hA, rows, Ctot, Ls, nL, hA, hB, yscr, stats, st);
}

extern "C" void kernel_launch(void* const* d_in, const int* in_sizes, int n_in,
                              void* d_out, int out_size, void* d_ws, size_t ws_size,
                              hipStream_t stream) {
  (void)in_sizes; (void)n_in; (void)out_size;
  // --- input map: insertion-order flattening of setup_inputs() dict ---
  // 0: xyz | sa1 b0 @1,5,9 b1 @13,17,21 | sa2 b0 @25.. b1 @37.. |
  // sa3 b0 @49.. b1 @61.. | fp3 @73,77 fp2 @81,85 fp1 @89,93 | head @97 |
  // conv2_W @101 conv2_b @102   (per-layer dict order: W, b, gamma, beta)
  const float* xyz = (const float*)d_in[0];
  const float* conv2_W = (const float*)d_in[101];
  const float* conv2_b = (const float*)d_in[102];

  // --- workspace layout (float units, every block 128B aligned) ---
  float* w = (float*)d_ws;
  size_t off = 0;
  auto alloc = [&](size_t nf) {
    float* p = w + off;
    off += (nf + 31) & ~(size_t)31;
    return p;
  };
  auto allocH = [&](size_t nh) { return (_Float16*)alloc((nh + 1) / 2); };

  float* xyz_t = alloc((size_t)BATCH * NPTS * 3);
  float* nx1   = alloc((size_t)BATCH * 1024 * 3);
  float* nx2   = alloc((size_t)BATCH * 256 * 3);
  float* nx3   = alloc((size_t)BATCH * 64 * 3);
  int*   fpsbuf = (int*)alloc((size_t)BATCH * 1024);
  int*   idxbuf = (int*)alloc((size_t)BATCH * 1024 * 32);
  _Float16* l1p = allocH((size_t)BATCH * 1024 * 128);
  _Float16* l2p = allocH((size_t)BATCH * 256 * 256);
  _Float16* l3p = allocH((size_t)BATCH * 64 * 512);
  _Float16* l2n = allocH((size_t)BATCH * 256 * 256);
  _Float16* l1n = allocH((size_t)BATCH * 1024 * 128);
  float* stats  = alloc(1024);
  _Float16* wreg = allocH((size_t)1024 * 1024);            // staged weights
  float*    yscr = alloc((size_t)16 * 1024 * 1024);        // f32 GEMM out
  _Float16* hA   = allocH((size_t)16 * 1024 * 1024);       // fp16 ping
  _Float16* hB   = allocH((size_t)16 * 1024 * 1024);       // fp16 pong
  if (off * sizeof(float) > ws_size) return;  // workspace too small

  // --- stage all weights into zero-padded fp16 ---
  size_t woff = 0;
  auto stage = [&](int idx, int Klog, int Nlog) {
    LayerEx L;
    L.W = (const float*)d_in[idx + 0];
    L.b = (const float*)d_in[idx + 1];
    L.gamma = (const float*)d_in[idx + 2];
    L.beta  = (const float*)d_in[idx + 3];
    L.Nlog = Nlog; L.Npad = pad32(Nlog);
    L.Klog = Klog; L.Kpad = pad32(Klog);
    L.Wp = wreg + woff;
    size_t n = (size_t)L.Npad * L.Kpad;
    woff += (n + 63) & ~(size_t)63;
    pad_w_kernel<<<cdivu((long long)n, 256), 256, 0, stream>>>(
        L.W, L.Wp, L.Nlog, L.Npad, L.Klog, L.Kpad);
    return L;
  };
  LayerEx sa1a[3] = { stage(1, 3, 32),    stage(5, 32, 32),   stage(9, 32, 64)   };
  LayerEx sa1b[3] = { stage(13, 3, 32),   stage(17, 32, 48),  stage(21, 48, 64)  };
  LayerEx sa2a[3] = { stage(25, 131, 64), stage(29, 64, 64),  stage(33, 64, 128) };
  LayerEx sa2b[3] = { stage(37, 131, 64), stage(41, 64, 96),  stage(45, 96, 128) };
  LayerEx sa3a[3] = { stage(49, 259, 128), stage(53, 128, 128), stage(57, 128, 256) };
  LayerEx sa3b[3] = { stage(61, 259, 128), stage(65, 128, 192), stage(69, 192, 256) };
  LayerEx fp3L[2] = { stage(73, 768, 512), stage(77, 512, 256) };
  LayerEx fp2L[2] = { stage(81, 384, 256), stage(85, 256, 128) };
  LayerEx fp1L[2] = { stage(89, 128, 128), stage(93, 128, 128) };
  LayerEx headL[1] = { stage(97, 128, 128) };

  // --- xyz transpose ---
  transpose_xyz_kernel<<<cdivu((long long)BATCH * NPTS, 256), 256, 0, stream>>>(xyz, xyz_t);

  // --- SA1: N=4096 -> 1024 centers, in features = xyz only ---
  fps_kernel<<<BATCH, FPS_BLOCK, 0, stream>>>(xyz_t, fpsbuf, NPTS, 1024);
  gather_xyz_kernel<<<cdivu((long long)BATCH * 1024, 256), 256, 0, stream>>>(
      xyz_t, fpsbuf, nx1, NPTS, 1024);
  sa_branch(xyz_t, NPTS, nx1, 1024, nullptr, 0, 0.1f, 16, sa1a, 3,
            idxbuf, hA, hB, yscr, stats, l1p, 128, 0, stream);
  sa_branch(xyz_t, NPTS, nx1, 1024, nullptr, 0, 0.2f, 32, sa1b, 3,
            idxbuf, hA, hB, yscr, stats, l1p, 128, 64, stream);

  // --- SA2: 1024 -> 256 centers, in features = l1p (128ch fp16) ---
  fps_kernel<<<BATCH, FPS_BLOCK, 0, stream>>>(nx1, fpsbuf, 1024, 256);
  gather_xyz_kernel<<<cdivu((long long)BATCH * 256, 256), 256, 0, stream>>>(
      nx1, fpsbuf, nx2, 1024, 256);
  sa_branch(nx1, 1024, nx2, 256, l1p, 128, 0.2f, 16, sa2a, 3,
            idxbuf, hA, hB, yscr, stats, l2p, 256, 0, stream);
  sa_branch(nx1, 1024, nx2, 256, l1p, 128, 0.4f, 32, sa2b, 3,
            idxbuf, hA, hB, yscr, stats, l2p, 256, 128, stream);

  // --- SA3: 256 -> 64 centers, in features = l2p (256ch fp16) ---
  fps_kernel<<<BATCH, FPS_BLOCK, 0, stream>>>(nx2, fpsbuf, 256, 64);
  gather_xyz_kernel<<<cdivu((long long)BATCH * 64, 256), 256, 0, stream>>>(
      nx2, fpsbuf, nx3, 256, 64);
  sa_branch(nx2, 256, nx3, 64, l2p, 256, 0.4f, 16, sa3a, 3,
            idxbuf, hA, hB, yscr, stats, l3p, 512, 0, stream);
  sa_branch(nx2, 256, nx3, 64, l2p, 256, 0.8f, 32, sa3b, 3,
            idxbuf, hA, hB, yscr, stats, l3p, 512, 256, stream);

  // --- FP3: interp l3 -> l2 centers; feat = [l2p(256), interp(512)] ---
  {
    const _Float16* r = fp_stage(nx2, 256, nx3, 64, l2p, 256, l3p, 512,
                                 fp3L, 2, hA, hB, yscr, stats, stream);
    copy_ch_kernel<<<cdivu((long long)BATCH * 256 * 256, 256), 256, 0, stream>>>(
        r, 256, l2n, 256, 0, BATCH * 256);
  }
  // --- FP2: interp l2n -> l1 centers; feat = [l1p(128), interp(256)] ---
  {
    const _Float16* r = fp_stage(nx1, 1024, nx2, 256, l1p, 128, l2n, 256,
                                 fp2L, 2, hA, hB, yscr, stats, stream);
    copy_ch_kernel<<<cdivu((long long)BATCH * 1024 * 128, 256), 256, 0, stream>>>(
        r, 128, l1n, 128, 0, BATCH * 1024);
  }
  // --- FP1: interp l1n -> all N points; feat = interp(128) only ---
  const _Float16* l0 = fp_stage(xyz_t, NPTS, nx1, 1024, nullptr, 0, l1n, 128,
                                fp1L, 2, hA, hB, yscr, stats, stream);

  // --- head MLP 128->128 + final 13-class conv ---
  const _Float16* h = mlp_chain(l0, BATCH * NPTS, 128, headL, 1,
                                hA, hB, yscr, stats, stream);
  final_conv_kernel<<<cdivu((long long)BATCH * NPTS * NCLS, 256), 256, 0, stream>>>(
      h, conv2_W, conv2_b, (float*)d_out);
}
